// DestroyAgent_25984552141632
// MI455X (gfx1250) — compile-verified, matrix operand order checked by reading, not run on hardware
//
#include <hip/hip_runtime.h>
#include <hip/hip_bf16.h>

// ---------------------------------------------------------------------------
// Problem constants (from the reference)
// ---------------------------------------------------------------------------
#define B_GR    256
#define K_SETS  256
#define D_SZ    3
#define N_PER   55
#define N_NODES (B_GR * N_PER)      // 14080  (= 880 * 16, exact 16-row tiles)
#define E_EDGES (N_NODES * 8)       // 112640
#define EMB     64
#define LAYERS  3

typedef __attribute__((ext_vector_type(16))) _Float16 v16h;
typedef __attribute__((ext_vector_type(8)))  float    v8f;

// ---------------------------------------------------------------------------
// Workspace layout (bytes, all 16B aligned)
// ---------------------------------------------------------------------------
#define WS_H     0
#define WS_AGG   (WS_H   + N_NODES * EMB * 4)                 // 3,604,480
#define WS_PACKW (WS_AGG + N_NODES * EMB * 4)                 // 7,208,960
#define PACKW_HALFS (LAYERS * 2 * 2 * 4 * 32 * 16)            // 24576 halfs
#define WS_XD    (WS_PACKW + PACKW_HALFS * 2)                 // 7,258,112
#define WS_PREG  (WS_XD + B_GR * N_PER * 4)                   // 7,314,432

// ---------------------------------------------------------------------------
// 0) generic zero fill
// ---------------------------------------------------------------------------
__global__ void zero_f32(float* __restrict__ p, int n) {
    int t = blockIdx.x * blockDim.x + threadIdx.x;
    if (t < n) p[t] = 0.0f;
}

// ---------------------------------------------------------------------------
// 1) pre-swizzle Wself/Wneigh (f32 [3,64,64]) into WMMA B fragments (f16).
// Fragment index: (((layer*2 + mat)*2 + kstep)*4 + ntile), per-lane 16 halfs.
// B layout (mirror of the ISA 16-bit A layout, column n per lane):
//   lane in [0,16):  n = lane,     K = k0 + {0..7, 16..23}
//   lane in [16,32): n = lane-16,  K = k0 + {8..15, 24..31}
// ---------------------------------------------------------------------------
__global__ void pack_weights(const float* __restrict__ Wself,
                             const float* __restrict__ Wneigh,
                             _Float16* __restrict__ packW) {
    int t = blockIdx.x * blockDim.x + threadIdx.x;
    if (t >= PACKW_HALFS) return;
    int e    = t & 15;
    int lane = (t >> 4) & 31;
    int nt   = (t >> 9) & 3;
    int ks   = (t >> 11) & 1;
    int mat  = (t >> 12) & 1;
    int l    = t >> 13;
    int lh   = lane >> 4;
    int k = ks * 32 + ((e >= 8) ? 16 : 0) + lh * 8 + (e & 7);
    int n = nt * 16 + (lane & 15);
    const float* W = (mat ? Wneigh : Wself) + l * EMB * EMB;
    packW[t] = (_Float16)W[k * EMB + n];
}

// ---------------------------------------------------------------------------
// 2) node coordinate embedding: h[n,f] = c0*We[0,f] + c1*We[1,f] + be[f]
// ---------------------------------------------------------------------------
__global__ void embed_nodes(const float* __restrict__ coords,
                            const float* __restrict__ We,
                            const float* __restrict__ be,
                            float* __restrict__ h) {
    int t = blockIdx.x * blockDim.x + threadIdx.x;   // N_NODES*EMB, exact grid
    int f = t & 63, n = t >> 6;
    h[t] = fmaf(coords[n * 2], We[f], fmaf(coords[n * 2 + 1], We[64 + f], be[f]));
}

// ---------------------------------------------------------------------------
// 3) segment-sum scatter: agg[dst[e]] += h[src[e]]   (f32 global atomics)
// one thread per (edge, 4-feature group) -> float4 load, 4 atomics
// ---------------------------------------------------------------------------
__global__ void scatter_edges(const float* __restrict__ h,
                              const int* __restrict__ src,
                              const int* __restrict__ dst,
                              float* __restrict__ agg) {
    int t = blockIdx.x * blockDim.x + threadIdx.x;   // E_EDGES*16, exact grid
    int e  = t >> 4;
    int fg = (t & 15) << 2;
    int s = src[e], d = dst[e];
    float4 v = *(const float4*)(h + s * EMB + fg);
    float* pd = agg + d * EMB + fg;
    atomicAdd(pd + 0, v.x);
    atomicAdd(pd + 1, v.y);
    atomicAdd(pd + 2, v.z);
    atomicAdd(pd + 3, v.w);
}

// ---------------------------------------------------------------------------
// 4) GNN layer GEMMs via WMMA:
//    h = relu(h@Wself[l] + agg@Wneigh[l] + b[l]) + h   (in place, tile-private)
// One wave -> one 16x64 output tile. 16 v_wmma_f32_16x16x32_f16 per wave.
// ---------------------------------------------------------------------------
__global__ __launch_bounds__(128) void gnn_matmul(float* h,
                                                  const float* __restrict__ agg,
                                                  const _Float16* __restrict__ packW,
                                                  const float* __restrict__ gnn_b,
                                                  int layer) {
    int lane = threadIdx.x & 31;
    int wave = threadIdx.x >> 5;
    int tile = blockIdx.x * 4 + wave;          // 0..879
    int m    = lane & 15;
    int lh   = lane >> 4;
    int row  = tile * 16 + m;

    const float* ph = h   + row * EMB;
    const float* pg = agg + row * EMB;

    // A fragments (16-bit A 16x32 layout): per lane, K = k0 + lh*8 + {0..7}
    // in elems 0..7 and K = k0 + 16 + lh*8 + {0..7} in elems 8..15.
    v16h aH[2], aG[2];
#pragma unroll
    for (int ks = 0; ks < 2; ++ks) {
        int kb = ks * 32 + lh * 8;
#pragma unroll
        for (int i = 0; i < 8; ++i) {
            aH[ks][i]     = (_Float16)ph[kb + i];
            aH[ks][8 + i] = (_Float16)ph[kb + 16 + i];
            aG[ks][i]     = (_Float16)pg[kb + i];
            aG[ks][8 + i] = (_Float16)pg[kb + 16 + i];
        }
    }

    const v16h* frag = (const v16h*)packW;
    int base = layer * 4;                      // (layer,mat=0) fragment base
#pragma unroll
    for (int nt = 0; nt < 4; ++nt) {
        v8f c = {};
#pragma unroll
        for (int ks = 0; ks < 2; ++ks) {
            v16h bS = frag[((base + ks) * 4 + nt) * 32 + lane];        // Wself
            c = __builtin_amdgcn_wmma_f32_16x16x32_f16(
                    false, aH[ks], false, bS, (short)0, c, false, false);
            v16h bN = frag[((base + 2 + ks) * 4 + nt) * 32 + lane];    // Wneigh
            c = __builtin_amdgcn_wmma_f32_16x16x32_f16(
                    false, aG[ks], false, bN, (short)0, c, false, false);
        }
        // epilogue: C/D layout -> elem r is (M = r + 8*lh, N = lane&15)
        int col   = nt * 16 + m;
        float bias = gnn_b[layer * EMB + col];
#pragma unroll
        for (int r = 0; r < 8; ++r) {
            int orow  = tile * 16 + r + 8 * lh;
            float pre = c[r] + bias;
            float res = h[orow * EMB + col];
            h[orow * EMB + col] = fmaxf(pre, 0.0f) + res;
        }
    }
}

// ---------------------------------------------------------------------------
// 5) per-(graph,node) MLP heads.
//    xg = MLP(h_gnn[b,n])  -> atomically folded into pre_g[b] via Wp[n]
//    xd = MLP(embed(coords[b,n])) -> x_d[b,n]   (only 55 distinct ids/graph)
// ---------------------------------------------------------------------------
__device__ __forceinline__ float mlp_tail(const float* l1,
                                          const float* sW2, const float* sb2,
                                          const float* sW3, float b3v) {
    float l2[32];
#pragma unroll
    for (int j = 0; j < 32; ++j) l2[j] = sb2[j];
#pragma unroll
    for (int i = 0; i < 32; ++i) {
        float x = l1[i] > 0.0f ? l1[i] : 0.01f * l1[i];   // leaky relu
#pragma unroll
        for (int j = 0; j < 32; ++j) l2[j] = fmaf(x, sW2[i * 32 + j], l2[j]);
    }
    float a = b3v;
#pragma unroll
    for (int i = 0; i < 32; ++i) {
        float x = l2[i] > 0.0f ? l2[i] : 0.01f * l2[i];
        a = fmaf(x, sW3[i], a);
    }
    return a;
}

__global__ __launch_bounds__(256) void mlp_heads(const float* __restrict__ h,
                                                 const float* __restrict__ coords,
                                                 const float* __restrict__ We,
                                                 const float* __restrict__ be,
                                                 const float* __restrict__ W1,
                                                 const float* __restrict__ b1,
                                                 const float* __restrict__ W2,
                                                 const float* __restrict__ b2,
                                                 const float* __restrict__ W3,
                                                 const float* __restrict__ b3,
                                                 const float* __restrict__ Wp,
                                                 float* __restrict__ x_d,
                                                 float* __restrict__ pre_g) {
    __shared__ float sW1[64 * 32];
    __shared__ float sW2[32 * 32];
    __shared__ float sb1[32], sb2[32], sW3[32];
    int tid = threadIdx.x;
    int r   = blockIdx.x * 256 + tid;          // exact: 55 blocks * 256 = 14080

    // hide LDS fill behind a global prefetch of this thread's GNN row
    __builtin_prefetch(h + r * EMB, 0, 0);

    for (int i = tid; i < 64 * 32; i += 256) sW1[i] = W1[i];
    for (int i = tid; i < 32 * 32; i += 256) sW2[i] = W2[i];
    if (tid < 32) { sb1[tid] = b1[tid]; sb2[tid] = b2[tid]; sW3[tid] = W3[tid]; }
    __syncthreads();

    int b = r / N_PER;
    int n = r - b * N_PER;
    float c0 = coords[r * 2], c1 = coords[r * 2 + 1];
    float b3v = b3[0];

    // ---- graph-embedding head (input: GNN output row) ----
    float l1[32];
#pragma unroll
    for (int j = 0; j < 32; ++j) l1[j] = sb1[j];
    for (int f = 0; f < 64; ++f) {
        float x = h[r * EMB + f];
#pragma unroll
        for (int j = 0; j < 32; ++j) l1[j] = fmaf(x, sW1[f * 32 + j], l1[j]);
    }
    float xg = mlp_tail(l1, sW2, sb2, sW3, b3v);

    // ---- destroy head (input: raw coord embedding, recomputed) ----
#pragma unroll
    for (int j = 0; j < 32; ++j) l1[j] = sb1[j];
    for (int f = 0; f < 64; ++f) {
        float x = fmaf(c0, We[f], fmaf(c1, We[64 + f], be[f]));
#pragma unroll
        for (int j = 0; j < 32; ++j) l1[j] = fmaf(x, sW1[f * 32 + j], l1[j]);
    }
    float xd = mlp_tail(l1, sW2, sb2, sW3, b3v);

    x_d[r] = xd;
    atomicAdd(&pre_g[b], xg * Wp[n]);
}

// ---------------------------------------------------------------------------
// 6) final: out[b,k] = pre_g[b] + bp + sum_d x_d[b, ids[b,k,d]] * Wp[55+d]
// ---------------------------------------------------------------------------
__global__ void final_out(const int* __restrict__ ids,
                          const float* __restrict__ x_d,
                          const float* __restrict__ pre_g,
                          const float* __restrict__ Wp,
                          const float* __restrict__ bp,
                          float* __restrict__ out) {
    int t = blockIdx.x * blockDim.x + threadIdx.x;   // B*K, exact grid
    int b = t >> 8;                                  // K = 256
    const int* id = ids + t * D_SZ;
    const float* xb = x_d + b * N_PER;
    float acc = pre_g[b] + bp[0];
    acc = fmaf(xb[id[0]], Wp[N_PER + 0], acc);
    acc = fmaf(xb[id[1]], Wp[N_PER + 1], acc);
    acc = fmaf(xb[id[2]], Wp[N_PER + 2], acc);
    out[t] = acc;
}

// ---------------------------------------------------------------------------
// launcher
// ---------------------------------------------------------------------------
extern "C" void kernel_launch(void* const* d_in, const int* in_sizes, int n_in,
                              void* d_out, int out_size, void* d_ws, size_t ws_size,
                              hipStream_t stream) {
    const float* coords      = (const float*)d_in[0];
    const int*   src         = (const int*)  d_in[1];
    const int*   dst         = (const int*)  d_in[2];
    const int*   destroy_ids = (const int*)  d_in[3];
    const float* W_embed     = (const float*)d_in[4];
    const float* b_embed     = (const float*)d_in[5];
    const float* Wself       = (const float*)d_in[6];
    const float* Wneigh      = (const float*)d_in[7];
    const float* gnn_b       = (const float*)d_in[8];
    const float* W1          = (const float*)d_in[9];
    const float* b1          = (const float*)d_in[10];
    const float* W2          = (const float*)d_in[11];
    const float* b2          = (const float*)d_in[12];
    const float* W3          = (const float*)d_in[13];
    const float* b3          = (const float*)d_in[14];
    const float* Wp          = (const float*)d_in[15];
    const float* bp          = (const float*)d_in[16];
    float* out = (float*)d_out;

    char* ws = (char*)d_ws;
    float*     h     = (float*)    (ws + WS_H);
    float*     agg   = (float*)    (ws + WS_AGG);
    _Float16*  packW = (_Float16*) (ws + WS_PACKW);
    float*     x_d   = (float*)    (ws + WS_XD);
    float*     pre_g = (float*)    (ws + WS_PREG);

    pack_weights<<<PACKW_HALFS / 256, 256, 0, stream>>>(Wself, Wneigh, packW);
    embed_nodes<<<(N_NODES * EMB) / 256, 256, 0, stream>>>(coords, W_embed, b_embed, h);

    for (int l = 0; l < LAYERS; ++l) {
        zero_f32<<<(N_NODES * EMB + 255) / 256, 256, 0, stream>>>(agg, N_NODES * EMB);
        scatter_edges<<<(E_EDGES * 16) / 256, 256, 0, stream>>>(h, src, dst, agg);
        gnn_matmul<<<220, 128, 0, stream>>>(h, agg, packW, gnn_b, l);
    }

    zero_f32<<<1, 256, 0, stream>>>(pre_g, B_GR);
    mlp_heads<<<N_NODES / 256, 256, 0, stream>>>(h, coords, W_embed, b_embed,
                                                 W1, b1, W2, b2, W3, b3, Wp,
                                                 x_d, pre_g);
    final_out<<<(B_GR * K_SETS) / 256, 256, 0, stream>>>(destroy_ids, x_d, pre_g,
                                                         Wp, bp, out);
}
